// GraphResolutionAttention_90494960927445
// MI455X (gfx1250) — matrix operand order repaired and checked by reading, not compiled, not run on hardware
//
#include <hip/hip_runtime.h>

typedef __attribute__((ext_vector_type(16))) _Float16 v16h;
typedef __attribute__((ext_vector_type(8)))  _Float16 v8h;
typedef __attribute__((ext_vector_type(8)))  float    v8f;

constexpr int kN  = 8192;
constexpr int kF  = 512;
constexpr int kFO = 64;
constexpr int kH  = 3;
constexpr int kJC = 128;          // j-chunk per flash iteration
constexpr float kNEG = -1e10f;

// ---------------------------------------------------------------------------
// Kernel 1: h = X @ W (exact f32), then e_self = h@a_self, e_neigh = h@a_neigh,
// and hT (f16, 64 x 8192, transposed so WMMA B-fragments are contiguous in K).
// One block per row i, 64 threads = one output feature each.
// ---------------------------------------------------------------------------
__global__ __launch_bounds__(64)
void k_linear(const float* __restrict__ X, const float* __restrict__ W,
              const float* __restrict__ a_self, const float* __restrict__ a_neigh,
              _Float16* __restrict__ hT, float* __restrict__ es, float* __restrict__ en)
{
    __shared__ float Xs[64];
    __shared__ float red1[64];
    __shared__ float red2[64];
    const int i = blockIdx.x;
    const int c = threadIdx.x;

    float acc = 0.f;
    for (int k0 = 0; k0 < kF; k0 += 64) {
        __syncthreads();
        Xs[c] = X[(size_t)i * kF + k0 + c];
        __syncthreads();
#pragma unroll
        for (int kk = 0; kk < 64; ++kk)
            acc = fmaf(Xs[kk], W[(size_t)(k0 + kk) * kFO + c], acc);
    }
    hT[(size_t)c * kN + i] = (_Float16)acc;
    red1[c] = acc * a_self[c];
    red2[c] = acc * a_neigh[c];
    __syncthreads();
    for (int s = 32; s > 0; s >>= 1) {
        if (c < s) { red1[c] += red1[c + s]; red2[c] += red2[c + s]; }
        __syncthreads();
    }
    if (c == 0) { es[i] = red1[0]; en[i] = red2[0]; }
}

// ---------------------------------------------------------------------------
// Kernel 2: fused mask + online softmax + (attn @ h) via WMMA.
// One block per 16-row tile; 128 threads = 4 waves; wave w owns output
// columns [16w, 16w+16). A is streamed exactly once (768 MB, non-temporal so
// it does not evict the L2-resident hT) -> HBM bandwidth bound (~33 us floor).
// ---------------------------------------------------------------------------
__global__ __launch_bounds__(128)
void k_attn(const float* __restrict__ A, const float* __restrict__ r,
            const _Float16* __restrict__ hT,
            const float* __restrict__ es_g, const float* __restrict__ en_g,
            float* __restrict__ out)
{
    __shared__ __align__(16) float    P32[16 * kJC];   // logits / probs (f32)
    __shared__ __align__(16) _Float16 Ph [16 * kJC];   // probs (f16) for WMMA A
    __shared__ float red[16 * 8];
    __shared__ float Mrow[16], Srow[16], Scl[16], ESs[16];

    const int tid  = threadIdx.x;
    const int lane = tid & 31;
    const int w    = tid >> 5;      // wave 0..3
    const int hh   = lane >> 4;     // lane half (0/1)
    const int nn   = lane & 15;
    const int i0   = blockIdx.x * 16;

    const float r0 = r[0], r1 = r[1], r2 = r[2];
    const float R  = r0 + r1 + r2;

    if (tid < 16) {
        Mrow[tid] = -INFINITY;
        Srow[tid] = 0.f;
        ESs[tid]  = es_g[i0 + tid];
    }
    __syncthreads();

    // hoist the 16 row biases into registers (avoid per-chunk ds reloads)
    float ess[16];
#pragma unroll
    for (int m = 0; m < 16; ++m) ess[m] = ESs[m];

    v8f acc = {};

    for (int j0 = 0; j0 < kN; j0 += kJC) {
        const int   j   = j0 + tid;
        const float enj = en_g[j];
        const float* Arow = A + ((size_t)i0 * kN + j) * kH;
        if (j0 + kJC < kN)
            __builtin_prefetch(A + ((size_t)i0 * kN + j + kJC) * kH);

        // ---- phase 1: logits (non-temporal streaming A reads) ----
        float L[16];
#pragma unroll 4
        for (int m = 0; m < 16; ++m) {
            const float* ap = Arow + (size_t)m * kN * kH;
            const float a0 = __builtin_nontemporal_load(ap);
            const float a1 = __builtin_nontemporal_load(ap + 1);
            const float a2 = __builtin_nontemporal_load(ap + 2);
            const float msk = (a0 == 0.f ? r0 : 0.f) +
                              (a1 == 0.f ? r1 : 0.f) +
                              (a2 == 0.f ? r2 : 0.f);
            const float x  = ess[m] + enj;
            const float d  = fmaxf(x, 0.f) + 0.2f * fminf(x, 0.f);  // leaky_relu 0.2
            const float lg = d * R + kNEG * msk;
            L[m] = lg;
            P32[m * kJC + tid] = lg;
        }
        __syncthreads();                                    // (a)

        // ---- phase 2: per-row chunk max, running max update ----
        {
            const int m = tid & 15, g = tid >> 4;
            float mx = -INFINITY;
#pragma unroll
            for (int q = 0; q < 16; ++q) mx = fmaxf(mx, P32[m * kJC + g * 16 + q]);
            red[m * 8 + g] = mx;
        }
        __syncthreads();                                    // (b)
        if (tid < 16) {
            float cm = red[tid * 8];
#pragma unroll
            for (int g = 1; g < 8; ++g) cm = fmaxf(cm, red[tid * 8 + g]);
            const float newM = fmaxf(Mrow[tid], cm);
            Scl[tid]  = __expf(Mrow[tid] - newM);
            Mrow[tid] = newM;
        }
        __syncthreads();                                    // (c)

        // ---- phase 3: exp, f16 probs, running sum update ----
#pragma unroll 4
        for (int m = 0; m < 16; ++m) {
            const float p = __expf(L[m] - Mrow[m]);
            Ph [m * kJC + tid] = (_Float16)p;
            P32[m * kJC + tid] = p;
        }
        __syncthreads();                                    // (d)
        {
            const int m = tid & 15, g = tid >> 4;
            float sm = 0.f;
#pragma unroll
            for (int q = 0; q < 16; ++q) sm += P32[m * kJC + g * 16 + q];
            red[m * 8 + g] = sm;
        }
        __syncthreads();                                    // (e)
        if (tid < 16) {
            float cs = 0.f;
#pragma unroll
            for (int g = 0; g < 8; ++g) cs += red[tid * 8 + g];
            Srow[tid] = Srow[tid] * Scl[tid] + cs;
        }
        __syncthreads();                                    // (f)

        // ---- phase 4: rescale accumulator, 4 WMMA K-steps of 32 ----
        // C/D layout: VGPR k holds row (k + 8*laneHalf), column (lane&15)+16w.
#pragma unroll
        for (int k = 0; k < 8; ++k) acc[k] *= Scl[k + 8 * hh];

#pragma unroll
        for (int ks = 0; ks < 4; ++ks) {
            const int kb = ks * 32;
            // A-frag (probs 16x32): lane row = nn; halves hold K {0..15}
            // (lanes 0-15) / {16..31} (lanes 16-31).
            v8h lo = *(const v8h*)&Ph[nn * kJC + kb + 16 * hh];
            v8h hi = *(const v8h*)&Ph[nn * kJC + kb + 16 * hh + 8];
            v16h afrag = __builtin_shufflevector(lo, hi,
                0, 1, 2, 3, 4, 5, 6, 7, 8, 9, 10, 11, 12, 13, 14, 15);
            // B-frag (h tile 32x16): column = 16w+nn, K contiguous from hT
            // (hT is 1 MB and stays L2-resident since A is non-temporal).
            const _Float16* bp = hT + (size_t)(16 * w + nn) * kN
                                    + (size_t)(j0 + kb + 16 * hh);
            v16h bfrag = *(const v16h*)bp;
            acc = __builtin_amdgcn_wmma_f32_16x16x32_f16(
                false, afrag, false, bfrag, (short)0, acc, false, false);
        }
    }

    // ---- epilogue: out = relu(O / S), streamed out non-temporally ----
    __syncthreads();
#pragma unroll
    for (int k = 0; k < 8; ++k) {
        const int row = k + 8 * hh;
        const float v = fmaxf(acc[k] / Srow[row], 0.f);
        __builtin_nontemporal_store(v, out + (size_t)(i0 + row) * kFO + (16 * w + nn));
    }
}

// ---------------------------------------------------------------------------
extern "C" void kernel_launch(void* const* d_in, const int* in_sizes, int n_in,
                              void* d_out, int out_size, void* d_ws, size_t ws_size,
                              hipStream_t stream)
{
    (void)in_sizes; (void)n_in; (void)out_size; (void)ws_size;
    const float* X       = (const float*)d_in[0];
    const float* A       = (const float*)d_in[1];
    const float* W       = (const float*)d_in[2];
    const float* a_self  = (const float*)d_in[3];
    const float* a_neigh = (const float*)d_in[4];
    const float* r       = (const float*)d_in[5];
    float* out = (float*)d_out;

    // workspace: hT (f16, 64x8192 = 1 MB) | e_self (32 KB) | e_neigh (32 KB)
    _Float16* hT = (_Float16*)d_ws;
    float* es = (float*)((char*)d_ws + (size_t)kFO * kN * sizeof(_Float16));
    float* en = es + kN;

    k_linear<<<kN, 64, 0, stream>>>(X, W, a_self, a_neigh, hT, es, en);
    k_attn<<<kN / 16, 128, 0, stream>>>(A, r, hT, es, en, out);
}
